// Crossview_Graph_Learning_9259949490767
// MI455X (gfx1250) — compile-verified, matrix operand order checked by reading, not run on hardware
//
#include <hip/hip_runtime.h>
#include <hip/hip_bf16.h>
#include <math.h>

// ---------------- problem constants ----------------
#define ND   3072          // N
#define TN   6144          // 2N
#define DD   128           // D
#define SQ   16            // S
#define EE   196608        // E
#define CC   100           // C
#define CPAD 112           // C padded to 16
#define HH   4
#define HD   32
#define SCALE_ATT 0.17677669529663687f   // 1/sqrt(32)
#define DELTA 0.1f
#define PRES  0.1f

typedef __attribute__((ext_vector_type(16))) __bf16 v16bf;
typedef __attribute__((ext_vector_type(8)))  float  v8f;

// k index for slot s (0..15) and lane-half (ISA 7.12.2 bf16 16x32 A layout):
// s<8 : VGPR0..3 -> k = s + 8*half ; s>=8 : VGPR4..7 -> k = 16 + (s-8) + 8*half
__device__ __forceinline__ int kmap(int s, int half) {
  return ((s & 8) << 1) + (s & 7) + (half << 3);
}

__device__ __forceinline__ v16bf ld_op_bf16(const __bf16* rowk0, int half) {
  v16bf r;
#pragma unroll
  for (int s = 0; s < 16; ++s) r[s] = rowk0[kmap(s, half)];
  return r;
}

__device__ __forceinline__ v8f wmma_bf16(v16bf a, v16bf b, v8f c) {
  return __builtin_amdgcn_wmma_f32_16x16x32_bf16(false, a, false, b, (short)0, c,
                                                 false, false);
}

// ---------------- small utility kernels ----------------
__global__ void fill_kernel(float* p, size_t n, float v) {
  size_t i = (size_t)blockIdx.x * 256 + threadIdx.x;
  size_t st = (size_t)gridDim.x * 256;
  for (; i < n; i += st) p[i] = v;
}

__global__ void rsqrt_kernel(float* dst, const float* src, int n) {
  int i = blockIdx.x * 256 + threadIdx.x;
  if (i < n) dst[i] = rsqrtf(src[i]);
}

// sums[0:128] = sum_n demand[n,15,:]; sums[128:256] = sum_n supply[n,15,:]
__global__ void reduce_last_kernel(const float* dem, const float* sup, float* sums) {
  int d = threadIdx.x;
  const float* p = blockIdx.x ? sup : dem;
  float a = 0.f;
  for (int n = 0; n < ND; ++n) a += p[(size_t)n * (SQ * DD) + 15 * DD + d];
  sums[blockIdx.x * DD + d] = a;
}

// cat (output 0) and query
__global__ void build_query_kernel(const float* skill, const float* sums,
                                   float* cat_out, float* query) {
  int idx = blockIdx.x * 256 + threadIdx.x;
  if (idx >= TN * DD) return;
  int i = idx >> 7, d = idx & 127;
  float sv = skill[(size_t)(i < ND ? i : i - ND) * DD + d];
  cat_out[idx] = sv;
  query[idx] = sv + (i < ND ? sums[d] : sums[DD + d]);
}

// Xc = concat([cat, seq_att], -1)  [TN,256]
__global__ void concat_xc_kernel(const float* skill, const float* seqatt, float* Xc) {
  int idx = blockIdx.x * 256 + threadIdx.x;
  if (idx >= TN * 256) return;
  int i = idx >> 8, k = idx & 255;
  Xc[idx] = (k < DD) ? skill[(size_t)(i < ND ? i : i - ND) * DD + k]
                     : seqatt[(size_t)i * DD + (k - DD)];
}

__global__ void copy_temps_kernel(const float* fused, float* tA, float* t2) {
  int idx = blockIdx.x * 256 + threadIdx.x;
  if (idx >= TN * DD) return;
  float v = fused[idx];
  tA[idx] = v;
  t2[idx] = v;
}

__global__ void init_out2_kernel(const float* dinv2, const float* xw, float* out2) {
  int idx = blockIdx.x * 256 + threadIdx.x;
  if (idx >= TN * DD) return;
  int i = idx >> 7;
  float di = dinv2[i];
  out2[idx] = di * di * xw[idx];
}

__global__ void combine_sparse_kernel(float* temp2, const float* out2, const float* bias) {
  int idx = blockIdx.x * 256 + threadIdx.x;
  if (idx >= TN * DD) return;
  int d = idx & 127;
  temp2[idx] = (1.f - PRES) * (out2[idx] + bias[d]) + PRES * temp2[idx];
}

__global__ void add_se_kernel(const float* tA, const float* t2, float* se, __bf16* seb) {
  int idx = blockIdx.x * 256 + threadIdx.x;
  if (idx >= TN * DD) return;
  float v = tA[idx] + t2[idx];
  se[idx] = v;
  seb[idx] = (__bf16)v;
}

__global__ void edge_deg_kernel(const int* dst, const float* w, float* deg2) {
  int e = blockIdx.x * 256 + threadIdx.x;
  if (e < EE) atomicAdd(&deg2[dst[e]], w[e]);
}

__global__ void scatter_gcn_kernel(const int* src, const int* dst, const float* w,
                                   const float* dinv2, const float* xw, float* out2) {
  int gid = blockIdx.x * 256 + threadIdx.x;
  int e = gid >> 5;
  if (e >= EE) return;
  int d = (gid & 31) * 4;
  int s = src[e], t = dst[e];
  float c = dinv2[s] * dinv2[t] * w[e];
  const float4 x = *(const float4*)(xw + (size_t)s * DD + d);
  float* o = out2 + (size_t)t * DD + d;
  atomicAdd(o + 0, c * x.x);
  atomicAdd(o + 1, c * x.y);
  atomicAdd(o + 2, c * x.z);
  atomicAdd(o + 3, c * x.w);
}

__global__ void scatterA_kernel(const int* src, const int* dst, const float* w, float* A) {
  int e = blockIdx.x * 256 + threadIdx.x;
  if (e < EE) atomicAdd(A + (size_t)src[e] * TN + dst[e], w[e]);
}

__global__ void final_loss_kernel(const float* acc, float* out) {
  // acc[0] = ent sum over rows, acc[1] = link sum of squares
  out[0] = sqrtf(acc[1]) / ((float)TN * (float)TN) + acc[0] / (float)TN;
}

// ---------------- generic GEMM (X [M,K] x W -> [M,Nout]) ----------------
// transB=1: W is [Nout,K] (torch Linear, Y = X @ W^T); transB=0: W is [K,wNout].
// B operands preconverted to bf16 and held in registers (one n-tile per wave).
struct GemmArgs {
  const float* X; int ldx;
  const float* W; int transB; int wNout;
  const float* bias;
  int M, Nout;
  float* Y; __bf16* Yb; int ldy;
  const float* P0; const float* P1; __bf16* Yb2;
};

template <int MODE, int KV>
__global__ __launch_bounds__(256) void gemm_kernel(GemmArgs g) {
  __shared__ __bf16 At[16 * (KV + 8)];
  const int AST = KV + 8;
  const int tid = threadIdx.x, lane = tid & 31, wid = tid >> 5;
  const int half = lane >> 4, nn = lane & 15;
  const int ntiles = g.Nout >> 4;
  const int tiles = g.M >> 4;

  v16bf breg[KV / 32];
  if (wid < ntiles) {
    const int n0 = wid << 4;
#pragma unroll
    for (int c = 0; c < KV / 32; ++c) {
      v16bf b;
#pragma unroll
      for (int s = 0; s < 16; ++s) {
        int k = c * 32 + kmap(s, half);
        float wv;
        if (g.transB) wv = g.W[(size_t)(n0 + nn) * KV + k];
        else          wv = (n0 + nn < g.wNout) ? g.W[(size_t)k * g.wNout + (n0 + nn)] : 0.f;
        b[s] = (__bf16)wv;
      }
      breg[c] = b;
    }
  }

  for (int it = blockIdx.x; it < tiles; it += gridDim.x) {
    __syncthreads();
    for (int idx = tid; idx < 16 * KV; idx += 256) {
      int r = idx / KV, k = idx - r * KV;
      At[r * AST + k] = (__bf16)g.X[(size_t)(it * 16 + r) * g.ldx + k];
    }
    __syncthreads();
    if (wid < ntiles) {
      const int n0 = wid << 4;
      v8f acc = (v8f)0.f;
      const __bf16* arow = At + nn * AST;
#pragma unroll
      for (int c = 0; c < KV / 32; ++c)
        acc = wmma_bf16(ld_op_bf16(arow + c * 32, half), breg[c], acc);
#pragma unroll
      for (int r = 0; r < 8; ++r) {
        int row = it * 16 + r + (half << 3);
        int col = n0 + nn;
        float v = acc[r];
        if (MODE == 0) {
          if (g.bias) v += g.bias[col];
          g.Y[(size_t)row * g.ldy + col] = v;
        } else if (MODE == 1) {
          v += g.bias[col];
          g.Yb[(size_t)row * g.ldy + col] = (__bf16)v;
        } else if (MODE == 2) {
          v += g.bias[col];
          g.Y[(size_t)row * g.ldy + col] = v;
          g.Yb[(size_t)row * g.ldy + col]  = (__bf16)tanhf(g.P0[0] * v);
          g.Yb2[(size_t)row * g.ldy + col] = (__bf16)tanhf(g.P1[0] * v);
        } else if (MODE == 3) {
          float y = g.P0[row] * v;
          g.Y[(size_t)row * g.ldy + col] = y;
          g.Yb[(size_t)row * g.ldy + col] = (__bf16)y;
        } else if (MODE == 4) {
          g.Y[(size_t)row * g.ldy + col] = (col < g.wNout) ? (v + g.bias[col]) : -1e30f;
        } else if (MODE == 5) {
          v += g.bias[col];
          g.Y[(size_t)row * g.ldy + col] = 2.f * g.P0[(size_t)row * DD + col] + v;
        }
      }
    }
  }
}

template <int MODE, int KV>
static void gemm(hipStream_t st, const float* X, int ldx, const float* W, int transB,
                 int wNout, const float* bias, int M, int Nout, float* Y, __bf16* Yb,
                 int ldy, const float* P0 = nullptr, const float* P1 = nullptr,
                 __bf16* Yb2 = nullptr) {
  GemmArgs g{X, ldx, W, transB, wNout, bias, M, Nout, Y, Yb, ldy, P0, P1, Yb2};
  int tiles = M >> 4;
  int grid = tiles < 2048 ? tiles : 2048;
  gemm_kernel<MODE, KV><<<grid, 256, 0, st>>>(g);
}

// ---------------- per-node attention (1 query x 16 keys, 4 heads) ----------------
__global__ __launch_bounds__(128) void attn1_kernel(const float* Q, const __bf16* Kb,
                                                    const __bf16* Vb, float* O) {
  int i = blockIdx.x;
  int lane = threadIdx.x & 31, h = threadIdx.x >> 5;
  int d = h * HD + lane;
  float q = Q[(size_t)i * DD + d];
  float sc[SQ];
#pragma unroll
  for (int t = 0; t < SQ; ++t) {
    float p = q * (float)Kb[(size_t)(i * SQ + t) * DD + d];
#pragma unroll
    for (int off = 16; off >= 1; off >>= 1) p += __shfl_xor(p, off, 32);
    sc[t] = p * SCALE_ATT;
  }
  float m = sc[0];
#pragma unroll
  for (int t = 1; t < SQ; ++t) m = fmaxf(m, sc[t]);
  float l = 0.f, acc = 0.f;
#pragma unroll
  for (int t = 0; t < SQ; ++t) {
    float e = __expf(sc[t] - m);
    l += e;
    acc += e * (float)Vb[(size_t)(i * SQ + t) * DD + d];
  }
  O[(size_t)i * DD + d] = acc / l;
}

// 6144 queries x 100 keys, 4 heads, online softmax
__global__ __launch_bounds__(256) void attn2_kernel(const float* Q2, const float* Kp,
                                                    const float* Vp, float* O2) {
  int lane = threadIdx.x & 31, wid = threadIdx.x >> 5;
  int h = wid & 3;
  int q0 = blockIdx.x * 32;
  for (int qi = (wid >> 2); qi < 32; qi += 2) {
    int q = q0 + qi;
    int d = h * HD + lane;
    float qv = Q2[(size_t)q * DD + d];
    float m = -1e30f, l = 0.f, acc = 0.f;
    for (int t = 0; t < CC; ++t) {
      float p = qv * Kp[t * DD + d];
#pragma unroll
      for (int off = 16; off >= 1; off >>= 1) p += __shfl_xor(p, off, 32);
      p *= SCALE_ATT;
      float mn = fmaxf(m, p);
      float sc = __expf(m - mn);
      float e = __expf(p - mn);
      l = l * sc + e;
      acc = acc * sc + e * Vp[t * DD + d];
      m = mn;
    }
    O2[(size_t)q * DD + d] = acc / l;
  }
}

// ---------------- scores: relu(s1 s2^T - s2 s1^T) + row max ----------------
__global__ __launch_bounds__(256) void scores_kernel(const __bf16* s1b, const __bf16* s2b,
                                                     float* predg, float* rowmax) {
  __shared__ __bf16 A1[16 * 136];
  __shared__ __bf16 A2[16 * 136];
  __shared__ float redbuf[8 * 16];
  int tid = threadIdx.x, lane = tid & 31, wid = tid >> 5;
  int i0 = blockIdx.x * 16;
  for (int idx = tid; idx < 16 * DD; idx += 256) {
    int r = idx >> 7, k = idx & 127;
    A1[r * 136 + k] = s1b[(size_t)(i0 + r) * DD + k];
    A2[r * 136 + k] = s2b[(size_t)(i0 + r) * DD + k];
  }
  __syncthreads();
  int half = lane >> 4, mn = lane & 15;
  const __bf16* a1row = A1 + mn * 136;
  const __bf16* a2row = A2 + mn * 136;
  float rmax[8];
#pragma unroll
  for (int r = 0; r < 8; ++r) rmax[r] = 0.f;

  for (int jt = wid; jt < TN / 16; jt += 8) {
    int j0 = jt * 16;
    const __bf16* b2row = s2b + (size_t)(j0 + mn) * DD;
    const __bf16* b1row = s1b + (size_t)(j0 + mn) * DD;
    v8f acc1 = (v8f)0.f, acc2 = (v8f)0.f;
#pragma unroll
    for (int k0 = 0; k0 < DD; k0 += 32) {
      acc1 = wmma_bf16(ld_op_bf16(a1row + k0, half), ld_op_bf16(b2row + k0, half), acc1);
      acc2 = wmma_bf16(ld_op_bf16(a2row + k0, half), ld_op_bf16(b1row + k0, half), acc2);
    }
#pragma unroll
    for (int r = 0; r < 8; ++r) {
      float v = fmaxf(acc1[r] - acc2[r], 0.f);
      int row = i0 + r + (half << 3);
      predg[(size_t)row * TN + j0 + mn] = v;
      rmax[r] = fmaxf(rmax[r], v);
    }
  }
#pragma unroll
  for (int r = 0; r < 8; ++r) {
    float v = rmax[r];
#pragma unroll
    for (int off = 1; off < 16; off <<= 1) v = fmaxf(v, __shfl_xor(v, off, 32));
    rmax[r] = v;
  }
  if ((lane & 15) == 0) {
#pragma unroll
    for (int r = 0; r < 8; ++r) redbuf[wid * 16 + r + (half << 3)] = rmax[r];
  }
  __syncthreads();
  if (tid < 16) {
    float v = redbuf[tid];
    for (int w = 1; w < 8; ++w) v = fmaxf(v, redbuf[w * 16 + tid]);
    rowmax[i0 + tid] = v;
  }
}

__global__ __launch_bounds__(256) void rowsum_kernel(const float* predg,
                                                     const float* rowmax, float* rowsum) {
  __shared__ float red[256];
  int i = blockIdx.x;
  float m = rowmax[i];
  float a = 0.f;
  for (int j = threadIdx.x; j < TN; j += 256) a += __expf(predg[(size_t)i * TN + j] - m);
  red[threadIdx.x] = a;
  __syncthreads();
  for (int s = 128; s > 0; s >>= 1) {
    if (threadIdx.x < s) red[threadIdx.x] += red[threadIdx.x + s];
    __syncthreads();
  }
  if (threadIdx.x == 0) rowsum[i] = red[0];
}

__global__ void normalize_kernel(float* predg, const float* rowmax, const float* rowsum) {
  size_t idx = (size_t)blockIdx.x * 256 + threadIdx.x;
  size_t st = (size_t)gridDim.x * 256;
  for (; idx < (size_t)TN * TN; idx += st) {
    int i = (int)(idx / TN);
    float v = __expf(predg[idx] - rowmax[i]) / rowsum[i] - DELTA;
    predg[idx] = fmaxf(v, 0.f);
  }
}

__global__ void colsum_kernel(const float* predg, float* deg) {
  int j = blockIdx.x * 256 + threadIdx.x;
  int r0 = blockIdx.y * 128;
  float a = 0.f;
  for (int r = 0; r < 128; ++r) a += predg[(size_t)(r0 + r) * TN + j];
  atomicAdd(&deg[j], a);
}

// ---------------- A^T * B kernel: C[t,d] = sum_k P[k,t] * Yb[k,d] ----------------
// EPI 0: GCN dense epilogue (self-loop + dinv scale + bias + residual mix, in place)
// EPI 1: plain store to Pool
template <int EPI>
__global__ __launch_bounds__(256) void atb_kernel(const float* P, int ldp, int Krows,
                                                  const __bf16* Yb, const float* dinv,
                                                  const float* y32, const float* bias,
                                                  float* temp, float* Pool) {
  __shared__ float Pt[32 * 17];
  __shared__ __bf16 Yt[32 * 136];
  int tid = threadIdx.x, lane = tid & 31, wid = tid >> 5;
  int t0 = blockIdx.x * 16;
  int half = lane >> 4, mn = lane & 15;
  v8f acc = (v8f)0.f;
  for (int s0 = 0; s0 < Krows; s0 += 32) {
    __syncthreads();
    for (int idx = tid; idx < 512; idx += 256) {
      int r = idx >> 4, c = idx & 15;
      Pt[r * 17 + c] = P[(size_t)(s0 + r) * ldp + t0 + c];
    }
    for (int idx = tid; idx < 32 * DD; idx += 256) {
      int r = idx >> 7, c = idx & 127;
      Yt[r * 136 + c] = Yb[(size_t)(s0 + r) * DD + c];
    }
    __syncthreads();
    v16bf a, b;
#pragma unroll
    for (int s = 0; s < 16; ++s) {
      int k = kmap(s, half);
      a[s] = (__bf16)Pt[k * 17 + mn];
      b[s] = Yt[k * 136 + wid * 16 + mn];
    }
    acc = wmma_bf16(a, b, acc);
  }
#pragma unroll
  for (int r = 0; r < 8; ++r) {
    int t = t0 + r + (half << 3);
    int d = wid * 16 + mn;
    if (EPI == 0) {
      float z = acc[r] + y32[(size_t)t * DD + d];          // + self loop (Ahat = P + I)
      float out = dinv[t] * z + bias[d];
      temp[(size_t)t * DD + d] = (1.f - PRES) * out + PRES * temp[(size_t)t * DD + d];
    } else {
      Pool[(size_t)t * DD + d] = acc[r];
    }
  }
}

// ---------------- softmax over pooled assignment + entropy ----------------
__global__ __launch_bounds__(128) void softmax_s_kernel(const float* logits, float* smat,
                                                        __bf16* sbf, float* ent_acc) {
  __shared__ float red[128];
  int i = blockIdx.x, t = threadIdx.x;
  float v = (t < CPAD) ? logits[(size_t)i * CPAD + t] : -1e30f;
  red[t] = v;
  __syncthreads();
  for (int s = 64; s > 0; s >>= 1) {
    if (t < s) red[t] = fmaxf(red[t], red[t + s]);
    __syncthreads();
  }
  float m = red[0];
  __syncthreads();
  float e = (t < CPAD) ? __expf(v - m) : 0.f;
  red[t] = e;
  __syncthreads();
  for (int s = 64; s > 0; s >>= 1) {
    if (t < s) red[t] += red[t + s];
    __syncthreads();
  }
  float sum = red[0];
  __syncthreads();
  float p = e / sum;
  if (t < CPAD) smat[(size_t)i * CPAD + t] = p;
  sbf[(size_t)i * DD + t] = (__bf16)((t < CPAD) ? p : 0.f);
  red[t] = (t < CPAD) ? -p * __logf(p + 1e-15f) : 0.f;
  __syncthreads();
  for (int s = 64; s > 0; s >>= 1) {
    if (t < s) red[t] += red[t + s];
    __syncthreads();
  }
  if (t == 0) atomicAdd(ent_acc, red[0]);
}

// ---------------- link loss: sum (A - s s^T)^2 ----------------
__global__ __launch_bounds__(256) void link_kernel(const __bf16* sbf, const float* A,
                                                   float* acc_ptr) {
  __shared__ __bf16 Arows[16 * 136];
  __shared__ float red[256];
  int tid = threadIdx.x, lane = tid & 31, wid = tid >> 5;
  int i0 = blockIdx.x * 16;
  for (int idx = tid; idx < 16 * DD; idx += 256) {
    int r = idx >> 7, k = idx & 127;
    Arows[r * 136 + k] = sbf[(size_t)(i0 + r) * DD + k];
  }
  __syncthreads();
  int half = lane >> 4, mn = lane & 15;
  const __bf16* arow = Arows + mn * 136;
  float lsum = 0.f;
  for (int jt = wid; jt < TN / 16; jt += 8) {
    int j0 = jt * 16;
    const __bf16* brow = sbf + (size_t)(j0 + mn) * DD;
    v8f acc = (v8f)0.f;
#pragma unroll
    for (int k0 = 0; k0 < DD; k0 += 32)
      acc = wmma_bf16(ld_op_bf16(arow + k0, half), ld_op_bf16(brow + k0, half), acc);
#pragma unroll
    for (int r = 0; r < 8; ++r) {
      int row = i0 + r + (half << 3);
      float d = A[(size_t)row * TN + j0 + mn] - acc[r];
      lsum += d * d;
    }
  }
  red[tid] = lsum;
  __syncthreads();
  for (int s = 128; s > 0; s >>= 1) {
    if (tid < s) red[tid] += red[tid + s];
    __syncthreads();
  }
  if (tid == 0) atomicAdd(acc_ptr, red[0]);
}

// ================= host side =================
#define ALIGN512(x) (((x) + 511) & ~((size_t)511))

extern "C" void kernel_launch(void* const* d_in, const int* in_sizes, int n_in,
                              void* d_out, int out_size, void* d_ws, size_t ws_size,
                              hipStream_t stream) {
  const float* demand = (const float*)d_in[0];
  const float* supply = (const float*)d_in[1];
  const float* skill  = (const float*)d_in[2];
  const int*   eidx   = (const int*)d_in[3];
  const float* eattr  = (const float*)d_in[4];
  const float* wfuse  = (const float*)d_in[5];
  const float* bfuse  = (const float*)d_in[6];
  const float* wi1    = (const float*)d_in[7];
  const float* bi1    = (const float*)d_in[8];
  const float* wo1    = (const float*)d_in[9];
  const float* bo1    = (const float*)d_in[10];
  const float* wi2    = (const float*)d_in[11];
  const float* bi2    = (const float*)d_in[12];
  const float* wo2    = (const float*)d_in[13];
  const float* bo2    = (const float*)d_in[14];
  const float* sender = (const float*)d_in[15];
  const float* recv   = (const float*)d_in[16];
  const float* W0     = (const float*)d_in[17];
  const float* b0     = (const float*)d_in[18];
  const float* W1     = (const float*)d_in[19];
  const float* b1     = (const float*)d_in[20];
  const float* Wp     = (const float*)d_in[21];
  const float* bp     = (const float*)d_in[22];
  const int* esrc = eidx;
  const int* edst = eidx + EE;

  float* out = (float*)d_out;
  float* out_cat   = out;                       // [TN,DD]
  float* out_skill = out + (size_t)TN * DD;     // [TN,DD]
  float* predg     = out + (size_t)2 * TN * DD; // [TN,TN]
  float* out_loss  = out + (size_t)2 * TN * DD + (size_t)TN * TN;

  char* ws = (char*)d_ws;
  size_t off = 0;
  auto carve = [&](size_t bytes) { size_t o = off; off = ALIGN512(off + bytes); return o; };
  float*  sums    = (float*)(ws + carve(256 * 4));
  float*  query   = (float*)(ws + carve((size_t)TN * DD * 4));
  float*  Q1      = (float*)(ws + carve((size_t)TN * DD * 4));
  __bf16* Kb      = (__bf16*)(ws + carve((size_t)TN * SQ * DD * 2));
  __bf16* Vb      = (__bf16*)(ws + carve((size_t)TN * SQ * DD * 2));
  float*  O1      = (float*)(ws + carve((size_t)TN * DD * 4));
  float*  seqatt  = (float*)(ws + carve((size_t)TN * DD * 4));
  float*  Xc      = (float*)(ws + carve((size_t)TN * 256 * 4));
  float*  fused   = (float*)(ws + carve((size_t)TN * DD * 4));
  __bf16* s1b     = (__bf16*)(ws + carve((size_t)TN * DD * 2));
  __bf16* s2b     = (__bf16*)(ws + carve((size_t)TN * DD * 2));
  float*  rowmax  = (float*)(ws + carve(TN * 4));
  float*  rowsum  = (float*)(ws + carve(TN * 4));
  float*  deg     = (float*)(ws + carve(TN * 4));
  float*  dinv    = (float*)(ws + carve(TN * 4));
  float*  deg2    = (float*)(ws + carve(TN * 4));
  float*  dinv2   = (float*)(ws + carve(TN * 4));
  float*  tempA   = (float*)(ws + carve((size_t)TN * DD * 4));
  float*  temp2   = (float*)(ws + carve((size_t)TN * DD * 4));
  float*  y32     = (float*)(ws + carve((size_t)TN * DD * 4));   // also sparse xw
  __bf16* y16     = (__bf16*)(ws + carve((size_t)TN * DD * 2));
  float*  out2    = (float*)(ws + carve((size_t)TN * DD * 4));
  float*  se      = (float*)(ws + carve((size_t)TN * DD * 4));
  __bf16* seb     = (__bf16*)(ws + carve((size_t)TN * DD * 2));
  float*  logits  = (float*)(ws + carve((size_t)TN * CPAD * 4));
  float*  smat    = (float*)(ws + carve((size_t)TN * CPAD * 4));
  __bf16* sbf     = (__bf16*)(ws + carve((size_t)TN * DD * 2));
  float*  pooled  = (float*)(ws + carve((size_t)CPAD * DD * 4));
  float*  Q2      = (float*)(ws + carve((size_t)TN * DD * 4));
  float*  Kp      = (float*)(ws + carve((size_t)CPAD * DD * 4));
  float*  Vp      = (float*)(ws + carve((size_t)CPAD * DD * 4));
  float*  O2      = (float*)(ws + carve((size_t)TN * DD * 4));
  float*  lacc    = (float*)(ws + carve(1024));                  // [0]=ent, [1]=link
  float*  Adense  = (float*)(ws + carve((size_t)TN * TN * 4));
  (void)ws_size; (void)in_sizes; (void)n_in; (void)out_size;

  // ---- init accumulators ----
  fill_kernel<<<1, 256, 0, stream>>>(lacc, 8, 0.f);
  fill_kernel<<<24, 256, 0, stream>>>(deg, TN, 1.f);   // self loop weight 1
  fill_kernel<<<24, 256, 0, stream>>>(deg2, TN, 1.f);
  fill_kernel<<<4096, 256, 0, stream>>>(Adense, (size_t)TN * TN, 0.f);

  // ---- MHA1 ----
  reduce_last_kernel<<<2, 128, 0, stream>>>(demand, supply, sums);
  build_query_kernel<<<(TN * DD) / 256, 256, 0, stream>>>(skill, sums, out_cat, query);
  gemm<0, 128>(stream, query, DD, wi1, 1, DD, bi1, TN, DD, Q1, nullptr, DD);
  gemm<1, 128>(stream, demand, DD, wi1 + DD * DD, 1, DD, bi1 + DD, ND * SQ, DD,
               nullptr, Kb, DD);
  gemm<1, 128>(stream, supply, DD, wi1 + DD * DD, 1, DD, bi1 + DD, ND * SQ, DD,
               nullptr, Kb + (size_t)ND * SQ * DD, DD);
  gemm<1, 128>(stream, demand, DD, wi1 + 2 * DD * DD, 1, DD, bi1 + 2 * DD, ND * SQ, DD,
               nullptr, Vb, DD);
  gemm<1, 128>(stream, supply, DD, wi1 + 2 * DD * DD, 1, DD, bi1 + 2 * DD, ND * SQ, DD,
               nullptr, Vb + (size_t)ND * SQ * DD, DD);
  attn1_kernel<<<TN, 128, 0, stream>>>(Q1, Kb, Vb, O1);
  gemm<0, 128>(stream, O1, DD, wo1, 1, DD, bo1, TN, DD, seqatt, nullptr, DD);

  // ---- fuse + tanh ----
  concat_xc_kernel<<<(TN * 256) / 256, 256, 0, stream>>>(skill, seqatt, Xc);
  gemm<2, 256>(stream, Xc, 256, wfuse, 0, DD, bfuse, TN, DD, fused, s1b, DD,
               sender, recv, s2b);

  // ---- graph update: scores -> softmax -> threshold (pred_g in d_out) ----
  scores_kernel<<<TN / 16, 256, 0, stream>>>(s1b, s2b, predg, rowmax);
  rowsum_kernel<<<TN, 256, 0, stream>>>(predg, rowmax, rowsum);
  normalize_kernel<<<4096, 256, 0, stream>>>(predg, rowmax, rowsum);
  colsum_kernel<<<dim3(TN / 256, 48), 256, 0, stream>>>(predg, deg);
  rsqrt_kernel<<<(TN + 255) / 256, 256, 0, stream>>>(dinv, deg, TN);

  // ---- GCN dense + sparse ----
  copy_temps_kernel<<<(TN * DD) / 256, 256, 0, stream>>>(fused, tempA, temp2);
  edge_deg_kernel<<<(EE + 255) / 256, 256, 0, stream>>>(edst, eattr, deg2);
  rsqrt_kernel<<<(TN + 255) / 256, 256, 0, stream>>>(dinv2, deg2, TN);
  for (int layer = 0; layer < 2; ++layer) {
    // dense: y = dinv * (temp @ W0[l]); temp = 0.9*(dinv*(P^T y + y) + b) + 0.1*temp
    gemm<3, 128>(stream, tempA, DD, W0 + layer * DD * DD, 0, DD, nullptr, TN, DD,
                 y32, y16, DD, dinv);
    atb_kernel<0><<<TN / 16, 256, 0, stream>>>(predg, TN, TN, y16, dinv, y32,
                                               b0 + layer * DD, tempA, nullptr);
    // sparse
    gemm<0, 128>(stream, temp2, DD, W1 + layer * DD * DD, 0, DD, nullptr, TN, DD,
                 y32, nullptr, DD);
    init_out2_kernel<<<(TN * DD) / 256, 256, 0, stream>>>(dinv2, y32, out2);
    scatter_gcn_kernel<<<(EE * 32) / 256, 256, 0, stream>>>(esrc, edst, eattr, dinv2,
                                                            y32, out2);
    combine_sparse_kernel<<<(TN * DD) / 256, 256, 0, stream>>>(temp2, out2,
                                                               b1 + layer * DD);
  }
  add_se_kernel<<<(TN * DD) / 256, 256, 0, stream>>>(tempA, temp2, se, seb);

  // ---- diff-pool ----
  gemm<4, 128>(stream, se, DD, Wp, 0, CC, bp, TN, CPAD, logits, nullptr, CPAD);
  softmax_s_kernel<<<TN, 128, 0, stream>>>(logits, smat, sbf, lacc + 0);
  atb_kernel<1><<<CPAD / 16, 256, 0, stream>>>(smat, CPAD, TN, seb, nullptr, nullptr,
                                               nullptr, nullptr, pooled);
  scatterA_kernel<<<(EE + 255) / 256, 256, 0, stream>>>(esrc, edst, eattr, Adense);
  link_kernel<<<TN / 16, 256, 0, stream>>>(sbf, Adense, lacc + 1);

  // ---- MHA2 over pooled clusters + residuals ----
  gemm<0, 128>(stream, se, DD, wi2, 1, DD, bi2, TN, DD, Q2, nullptr, DD);
  gemm<0, 128>(stream, pooled, DD, wi2 + DD * DD, 1, DD, bi2 + DD, CPAD, DD, Kp,
               nullptr, DD);
  gemm<0, 128>(stream, pooled, DD, wi2 + 2 * DD * DD, 1, DD, bi2 + 2 * DD, CPAD, DD,
               Vp, nullptr, DD);
  attn2_kernel<<<TN / 32, 256, 0, stream>>>(Q2, Kp, Vp, O2);
  // skill_out = 2*skill_embs + (O2 @ wo2^T + bo2)
  gemm<5, 128>(stream, O2, DD, wo2, 1, DD, bo2, TN, DD, out_skill, nullptr, DD, se);

  final_loss_kernel<<<1, 1, 0, stream>>>(lacc, out_loss);
}
